// TruthGPTFlashAttention_50964081934740
// MI455X (gfx1250) — matrix operand order repaired and checked by reading, not compiled, not run on hardware
//
#include <hip/hip_runtime.h>
#include <stdint.h>

typedef __bf16 bf16;
typedef __bf16 v8bf  __attribute__((ext_vector_type(8)));   // 16 bytes
typedef __bf16 v16bf __attribute__((ext_vector_type(16)));  // 32 bytes
typedef float  v8f   __attribute__((ext_vector_type(8)));
typedef unsigned int uint4v __attribute__((ext_vector_type(4)));
typedef int int8v  __attribute__((ext_vector_type(8)));
typedef int int4v  __attribute__((ext_vector_type(4)));

#define WMMA_BF16(a, b, c) \
  __builtin_amdgcn_wmma_f32_16x16x32_bf16(false, (a), false, (b), (short)0, (c), false, false)

// --- CDNA5 async global->LDS (ASYNCcnt) ------------------------------------
__device__ inline void async_ld_b128(unsigned lds_addr, const bf16* gaddr) {
  asm volatile("global_load_async_to_lds_b128 %0, %1, off"
               :: "v"(lds_addr), "v"(gaddr) : "memory");
}
__device__ inline void wait_async0() {
#if __has_builtin(__builtin_amdgcn_s_wait_asynccnt)
  __builtin_amdgcn_s_wait_asynccnt(0);
#else
  asm volatile("s_wait_asynccnt 0x0" ::: "memory");
#endif
}
__device__ inline unsigned lds_addr_of(const void* p) {
  return (unsigned)(unsigned long long)p;  // generic LDS addr: low 32 bits = LDS offset
}

// --- CDNA5 Tensor Data Mover (TENSORcnt) -----------------------------------
// 2D tile load: tile_x x tile_y elements (2-byte), row stride stride_x elems,
// LDS rows padded by 16B every 64B (matches LDT=40-half layout).
__device__ inline void tdm_load_tile_2d(unsigned lds_addr, const void* gaddr,
                                        unsigned tile_x, unsigned tile_y,
                                        unsigned tensor_x, unsigned tensor_y,
                                        unsigned stride_x) {
  unsigned long long ga = (unsigned long long)gaddr;
  unsigned ga_lo = __builtin_amdgcn_readfirstlane((unsigned)(ga & 0xFFFFFFFFu));
  unsigned ga_hi = __builtin_amdgcn_readfirstlane((unsigned)(ga >> 32));
  unsigned lds   = __builtin_amdgcn_readfirstlane(lds_addr);
  uint4v g0;
  g0[0] = 1u;                                   // count=1, user descriptor
  g0[1] = lds;                                  // lds_addr (bytes)
  g0[2] = ga_lo;                                // global_addr[31:0]
  g0[3] = (ga_hi & 0x1FFFFFFu) | (2u << 30);    // global_addr[56:32] | type=2
  int8v g1;
  g1[0] = (int)((1u << 16) |                    // data_size = 2 bytes
                (1u << 20) |                    // pad_enable
                (3u << 22) |                    // pad_interval: 16 DWORDs (64B)
                (3u << 25));                    // pad_amount: 4 DWORDs (16B)
  g1[1] = (int)((tensor_x & 0xFFFFu) << 16);    // tensor_dim0[15:0]
  g1[2] = (int)((tensor_x >> 16) | ((tensor_y & 0xFFFFu) << 16));
  g1[3] = (int)((tensor_y >> 16) | (tile_x << 16));   // tile_dim0
  g1[4] = (int)(tile_y & 0xFFFFu);              // tile_dim1 (tile_dim2=0)
  g1[5] = (int)stride_x;                        // tensor_dim0_stride[31:0]
  g1[6] = 0;
  g1[7] = 0;
  int4v z4 = {0, 0, 0, 0};
#if __clang_major__ >= 23
  int8v z8 = {0, 0, 0, 0, 0, 0, 0, 0};
  __builtin_amdgcn_tensor_load_to_lds(g0, g1, z4, z4, z8, 0);
#else
  __builtin_amdgcn_tensor_load_to_lds(g0, g1, z4, z4, 0);
#endif
}
__device__ inline void wait_tensor0() {
  __builtin_amdgcn_s_wait_tensorcnt((short)0);
}

// ---------------------------------------------------------------------------
// Fragment loaders from row-major tiles (row stride ldk halves).
// A-matrix 16x32 bf16 (ISA 7.12.2): lanes 0-15: K=0..7 (v0-3), K=16..23 (v4-7);
// lanes 16-31: K=8..15, K=24..31.
__device__ inline v16bf load_frag_a(const bf16* base, int ldk, int lane) {
  int r = lane & 15, hh = lane >> 4;
  const bf16* p = base + r * ldk + hh * 8;
  union { v16bf v; v8bf h2[2]; } u;
  u.h2[0] = *(const v8bf*)(p);
  u.h2[1] = *(const v8bf*)(p + 16);
  return u.v;
}
// B-matrix 32x16 bf16: lane holds column n = lane&15; lanes 0-15: K=0..7,
// K=16..23; lanes 16-31: K=8..15, K=24..31.
__device__ inline v16bf load_frag_b(const bf16* base, int ldk, int lane) {
  int r = lane & 15, hh = lane >> 4;
  const bf16* p = base + r * ldk + hh * 8;
  union { v16bf v; v8bf h2[2]; } u;
  u.h2[0] = *(const v8bf*)(p);
  u.h2[1] = *(const v8bf*)(p + 16);
  return u.v;
}

// ---------------------------------------------------------------------------
__global__ void k_f32_to_bf16(const float* __restrict__ x, bf16* __restrict__ y, int n) {
  int i = blockIdx.x * blockDim.x + threadIdx.x;
  int stride = gridDim.x * blockDim.x;
  for (; i < n; i += stride) y[i] = (bf16)x[i];
}

// w: [K,N] f32 row-major -> wt: [N,K] bf16 row-major (transpose + convert)
__global__ void k_transpose_bf16(const float* __restrict__ w, bf16* __restrict__ wt,
                                 int K, int N) {
  int i = blockIdx.x * blockDim.x + threadIdx.x;
  int total = K * N;
  int stride = gridDim.x * blockDim.x;
  for (; i < total; i += stride) {
    int k = i / N, n = i - k * N;
    wt[(size_t)n * K + k] = (bf16)w[i];
  }
}

// ---------------------------------------------------------------------------
// GEMM: C[M,N] = A[M,K] * Bt[N,K]^T + bias. Tiles staged by the Tensor Data
// Mover (one descriptor per tile, wave-0 issues), double-buffered on TENSORcnt.
// mode 0: out = f32 row-major [M,N]; mode 1: out = bf16 head-major [B,NH,S,HD]
#define BM 128
#define BN 128
#define BK 32
#define LDT 40  // LDS row stride (halves): 80B = 64B data + 16B TDM pad

__global__ void __launch_bounds__(256) k_gemm_bf16(
    const bf16* __restrict__ A, const bf16* __restrict__ Bt,
    const float* __restrict__ bias, void* __restrict__ out,
    int M, int N, int K, int mode, int S, int NHp, int HDp) {
  __shared__ __align__(16) bf16 sA[2][BM * LDT];
  __shared__ __align__(16) bf16 sB[2][BN * LDT];
  int tid = threadIdx.x;
  int lane = tid & 31;
  int wave = tid >> 5;   // 8 waves
  int wm = wave >> 2;    // 0..1
  int wn = wave & 3;     // 0..3
  int m0 = blockIdx.y * BM;
  int n0 = blockIdx.x * BN;

  bool leader = (__builtin_amdgcn_readfirstlane(tid) < 32);  // scalar: wave 0
  const bf16* gA = A + (size_t)m0 * K;   // tile row origin; k0 advances
  const bf16* gB = Bt + (size_t)n0 * K;
  unsigned lA = lds_addr_of(&sA[0][0]);
  unsigned lB = lds_addr_of(&sB[0][0]);
  const unsigned bufHopA = BM * LDT * 2;  // bytes
  const unsigned bufHopB = BN * LDT * 2;

  v8f acc[4][2];
  for (int i = 0; i < 4; i++)
    for (int j = 0; j < 2; j++)
      for (int e = 0; e < 8; e++) acc[i][j][e] = 0.0f;

  const int nk = K / BK;
  if (leader) {  // prologue: stage tile 0 into buffer 0 via TDM
    tdm_load_tile_2d(lA, gA, BK, BM, K, M - m0, K);
    tdm_load_tile_2d(lB, gB, BK, BN, K, N - n0, K);
  }

  for (int ks = 0; ks < nk; ks++) {
    if (leader) wait_tensor0();
    __syncthreads();  // tile ks resident in buf ks&1; prior reads of other buf done
    if (leader && (ks + 1 < nk)) {
      size_t g = (size_t)(ks + 1) * BK;
      unsigned ob = (unsigned)((ks + 1) & 1);
      tdm_load_tile_2d(lA + ob * bufHopA, gA + g, BK, BM, K, M - m0, K);
      tdm_load_tile_2d(lB + ob * bufHopB, gB + g, BK, BN, K, N - n0, K);
    }
    const bf16* tA = sA[ks & 1];
    const bf16* tB = sB[ks & 1];
    v16bf aF[4], bF[2];
    for (int i = 0; i < 4; i++) aF[i] = load_frag_a(&tA[(wm * 64 + i * 16) * LDT], LDT, lane);
    for (int j = 0; j < 2; j++) bF[j] = load_frag_b(&tB[(wn * 32 + j * 16) * LDT], LDT, lane);
    for (int i = 0; i < 4; i++)
      for (int j = 0; j < 2; j++)
        acc[i][j] = WMMA_BF16(aF[i], bF[j], acc[i][j]);
  }

  // Epilogue. C layout: VGPR r, lanes 0-15: (M=r, N=lane); lanes 16-31: (M=8+r, N=lane-16)
  int cn = lane & 15;
  int hh = lane >> 4;
  for (int i = 0; i < 4; i++) {
    for (int j = 0; j < 2; j++) {
      int n = n0 + wn * 32 + j * 16 + cn;
      float bv = bias ? bias[n] : 0.0f;
      for (int r = 0; r < 8; r++) {
        int m = m0 + wm * 64 + i * 16 + r + 8 * hh;
        float v = acc[i][j][r] + bv;
        if (mode == 0) {
          ((float*)out)[(size_t)m * N + n] = v;
        } else {
          int bb2 = m / S, ss = m - bb2 * S;
          int head = n / HDp, d = n - head * HDp;
          ((bf16*)out)[(((size_t)bb2 * NHp + head) * S + ss) * (size_t)HDp + d] = (bf16)v;
        }
      }
    }
  }
}

// ---------------------------------------------------------------------------
// Flash attention: block = (b, h, 128 q rows); 8 waves x 16 q rows.
// Double-buffered KV tiles: K via async-to-LDS (contiguous), V via vector load
// + transpose scatter, prefetched mid-iteration to hide behind softmax.
#define AQ  128
#define AKV 64
#define LHD 72  // padded stride for 64-wide rows (144B, 16B aligned)

__global__ void __launch_bounds__(256) k_flash_attn(
    const bf16* __restrict__ Q, const bf16* __restrict__ Kb, const bf16* __restrict__ Vb,
    bf16* __restrict__ ctx, int Sd, int NHd, int HDd, float scale) {
  __shared__ __align__(16) bf16 sK[2][AKV * LHD];   // [n][d]
  __shared__ __align__(16) bf16 sV[2][64 * LHD];    // V^T tile [d][n]
  __shared__ __align__(16) bf16 sP[8 * 16 * LHD];   // per-wave P tiles [16][LHD]

  int tid = threadIdx.x, lane = tid & 31, wave = tid >> 5;
  int q0 = blockIdx.x * AQ;
  int head = blockIdx.y, bb = blockIdx.z;
  size_t headOff = (((size_t)bb * NHd + head) * Sd) * (size_t)HDd;
  const bf16* Qh = Q + headOff;
  const bf16* Kh = Kb + headOff;
  const bf16* Vh = Vb + headOff;

  int r = lane & 15, hh = lane >> 4;

  // Tile-load assignment: rows row0 and row0+32, fixed 16B column.
  int row0 = tid >> 3;           // 0..31
  int offk = (tid & 7) * 8;
  const bf16* gK0 = Kh + (size_t)row0 * HDd + offk;
  const bf16* gK1 = gK0 + (size_t)32 * HDd;
  const bf16* gV0 = Vh + (size_t)row0 * HDd + offk;
  const bf16* gV1 = gV0 + (size_t)32 * HDd;
  unsigned lK = lds_addr_of(&sK[0][row0 * LHD + offk]);
  const unsigned rowHopK = 32 * LHD * 2;
  const unsigned bufHopK = AKV * LHD * 2;

  // Preload Q fragments (A layout) straight from global: per-lane b128 loads
  v16bf qF[2];
  {
    const bf16* qp = Qh + (size_t)(q0 + wave * 16 + r) * HDd;
    for (int ks = 0; ks < 2; ks++) {
      const bf16* p = qp + ks * 32 + hh * 8;
      union { v16bf v; v8bf h2[2]; } u;
      u.h2[0] = *(const v8bf*)(p);
      u.h2[1] = *(const v8bf*)(p + 16);
      qF[ks] = u.v;
    }
  }

  v8f oacc[4];
  float mrow[8], lrow[8];
  for (int t = 0; t < 4; t++)
    for (int e = 0; e < 8; e++) oacc[t][e] = 0.0f;
  for (int e = 0; e < 8; e++) { mrow[e] = -1e30f; lrow[e] = 0.0f; }

  bf16* pw = &sP[wave * 16 * LHD];
  const int nkv = Sd / AKV;

  // Prologue: stage KV tile 0 into buffer 0
  async_ld_b128(lK, gK0);
  async_ld_b128(lK + rowHopK, gK1);
  {
    v8bf v0 = *(const v8bf*)(gV0);
    v8bf v1 = *(const v8bf*)(gV1);
    for (int e = 0; e < 8; e++) {
      sV[0][(offk + e) * LHD + row0] = v0[e];
      sV[0][(offk + e) * LHD + row0 + 32] = v1[e];
    }
  }

  for (int it = 0; it < nkv; it++) {
    int cb = it & 1, nb = (it + 1) & 1;
    wait_async0();
    __syncthreads();  // tile it resident; prior reads of other buffer done

    // Scores S = Q K^T (B operand = K rows)
    v8f sc[4];
    for (int t = 0; t < 4; t++)
      for (int e = 0; e < 8; e++) sc[t][e] = 0.0f;
    for (int t = 0; t < 4; t++) {
      v16bf kf = load_frag_b(&sK[cb][(t * 16) * LHD], LHD, lane);
      sc[t] = WMMA_BF16(qF[0], kf, sc[t]);
      kf = load_frag_b(&sK[cb][(t * 16) * LHD + 32], LHD, lane);
      sc[t] = WMMA_BF16(qF[1], kf, sc[t]);
    }
    // apply 1/sqrt(HD) once
    for (int t = 0; t < 4; t++)
      for (int e = 0; e < 8; e++) sc[t][e] *= scale;

    // Prefetch next KV tile while softmax runs
    if (it + 1 < nkv) {
      size_t g = (size_t)(it + 1) * AKV * HDd;
      async_ld_b128(lK + nb * bufHopK, gK0 + g);
      async_ld_b128(lK + nb * bufHopK + rowHopK, gK1 + g);
      v8bf v0 = *(const v8bf*)(gV0 + g);
      v8bf v1 = *(const v8bf*)(gV1 + g);
      for (int e = 0; e < 8; e++) {
        sV[nb][(offk + e) * LHD + row0] = v0[e];
        sV[nb][(offk + e) * LHD + row0 + 32] = v1[e];
      }
    }

    // Online softmax; row (e + 8*hh) lives across 16 lanes of this half-wave
    for (int e = 0; e < 8; e++) {
      float mx = fmaxf(fmaxf(sc[0][e], sc[1][e]), fmaxf(sc[2][e], sc[3][e]));
      for (int msk = 8; msk >= 1; msk >>= 1) mx = fmaxf(mx, __shfl_xor(mx, msk, 32));
      float mnew = fmaxf(mrow[e], mx);
      float al = __expf(mrow[e] - mnew);
      float sum = 0.0f;
      for (int t = 0; t < 4; t++) {
        float p = __expf(sc[t][e] - mnew);
        sc[t][e] = p;
        sum += p;
      }
      for (int msk = 8; msk >= 1; msk >>= 1) sum += __shfl_xor(sum, msk, 32);
      lrow[e] = lrow[e] * al + sum;
      mrow[e] = mnew;
      for (int t = 0; t < 4; t++) oacc[t][e] *= al;
    }

    // C-layout -> A-layout via per-wave LDS round trip (same-wave DS in-order)
    for (int t = 0; t < 4; t++)
      for (int e = 0; e < 8; e++)
        pw[(e + 8 * hh) * LHD + t * 16 + r] = (bf16)sc[t][e];
    __syncthreads();

    // ctx += P @ V   (B operand = sV rows = V columns d)
    v16bf pF0 = load_frag_a(pw, LHD, lane);       // n = 0..31
    v16bf pF1 = load_frag_a(pw + 32, LHD, lane);  // n = 32..63
    for (int t = 0; t < 4; t++) {
      v16bf vf = load_frag_b(&sV[cb][(t * 16) * LHD], LHD, lane);
      oacc[t] = WMMA_BF16(pF0, vf, oacc[t]);
      vf = load_frag_b(&sV[cb][(t * 16) * LHD + 32], LHD, lane);
      oacc[t] = WMMA_BF16(pF1, vf, oacc[t]);
    }
  }

  // Normalize and write ctx in [B*S, AHS] layout (col = head*HD + d)
  int AHSd = NHd * HDd;
  for (int t = 0; t < 4; t++) {
    for (int e = 0; e < 8; e++) {
      int s = q0 + wave * 16 + e + 8 * hh;
      float v = oacc[t][e] / lrow[e];
      int d = t * 16 + r;
      ctx[((size_t)bb * Sd + s) * AHSd + head * HDd + d] = (bf16)v;
    }
  }
}

// ---------------------------------------------------------------------------
extern "C" void kernel_launch(void* const* d_in, const int* in_sizes, int n_in,
                              void* d_out, int out_size, void* d_ws, size_t ws_size,
                              hipStream_t stream) {
  (void)in_sizes; (void)n_in; (void)out_size; (void)ws_size;
  const float* hs = (const float*)d_in[0];
  const float* wq = (const float*)d_in[1];
  const float* bq = (const float*)d_in[2];
  const float* wk = (const float*)d_in[3];
  const float* bk = (const float*)d_in[4];
  const float* wv = (const float*)d_in[5];
  const float* bv = (const float*)d_in[6];
  const float* wo = (const float*)d_in[7];
  const float* bo = (const float*)d_in[8];
  float* out = (float*)d_out;

  const int B = 2, S = 2048, H = 1024, NH = 16, HD = 64, AHS = NH * HD;
  const int M = B * S;

  size_t off = 0;
  char* base = (char*)d_ws;
  auto alloc = [&](size_t bytes) {
    char* p = base + off;
    off += (bytes + 255) & ~(size_t)255;
    return p;
  };
  bf16* Xbf = (bf16*)alloc((size_t)M * H * sizeof(bf16));
  bf16* WqT = (bf16*)alloc((size_t)H * AHS * sizeof(bf16));
  bf16* WkT = (bf16*)alloc((size_t)H * AHS * sizeof(bf16));
  bf16* WvT = (bf16*)alloc((size_t)H * AHS * sizeof(bf16));
  bf16* WoT = (bf16*)alloc((size_t)AHS * H * sizeof(bf16));
  bf16* Qh  = (bf16*)alloc((size_t)B * NH * S * HD * sizeof(bf16));
  bf16* Kh  = (bf16*)alloc((size_t)B * NH * S * HD * sizeof(bf16));
  bf16* Vh  = (bf16*)alloc((size_t)B * NH * S * HD * sizeof(bf16));
  bf16* Ctx = (bf16*)alloc((size_t)M * AHS * sizeof(bf16));

  // Convert / transpose to bf16
  k_f32_to_bf16<<<2048, 256, 0, stream>>>(hs, Xbf, M * H);
  k_transpose_bf16<<<1024, 256, 0, stream>>>(wq, WqT, H, AHS);
  k_transpose_bf16<<<1024, 256, 0, stream>>>(wk, WkT, H, AHS);
  k_transpose_bf16<<<1024, 256, 0, stream>>>(wv, WvT, H, AHS);
  k_transpose_bf16<<<1024, 256, 0, stream>>>(wo, WoT, AHS, H);

  // QKV projections (head-major scatter epilogue)
  dim3 gq(AHS / BN, M / BM);
  k_gemm_bf16<<<gq, 256, 0, stream>>>(Xbf, WqT, bq, Qh, M, AHS, H, 1, S, NH, HD);
  k_gemm_bf16<<<gq, 256, 0, stream>>>(Xbf, WkT, bk, Kh, M, AHS, H, 1, S, NH, HD);
  k_gemm_bf16<<<gq, 256, 0, stream>>>(Xbf, WvT, bv, Vh, M, AHS, H, 1, S, NH, HD);

  // Flash attention over heads
  dim3 ga(S / AQ, NH, B);
  k_flash_attn<<<ga, 256, 0, stream>>>(Qh, Kh, Vh, Ctx, S, NH, HD, 0.125f);

  // Output projection (f32 row-major + bias)
  dim3 go(H / BN, M / BM);
  k_gemm_bf16<<<go, 256, 0, stream>>>(Ctx, WoT, bo, out, M, H, AHS, 0, S, NH, HD);
}
